// IPAttnProcessor_27719718928953
// MI455X (gfx1250) — matrix operand order repaired and checked by reading, not compiled, not run on hardware
//
#include <hip/hip_runtime.h>
#include <hip/hip_bf16.h>

constexpr int kB = 8, kSQ = 4096, kC = 1280, kCAD = 2048, kH = 20, kD = 64;
constexpr int kNT = 4, kTXT = 77;
constexpr float kScale = 0.125f;   // 1/sqrt(D)
constexpr float kIpScale = 1.0f;

typedef __attribute__((ext_vector_type(16))) _Float16 v16h;
typedef __attribute__((ext_vector_type(8)))  float    v8f;

union Frag { v16h v; unsigned int u[8]; };
union HU { _Float16 h; unsigned short u; };

__device__ __forceinline__ unsigned short f2h(float f) {
  HU x; x.h = (_Float16)f; return x.u;
}
__device__ __forceinline__ unsigned int pack2(float a, float b) {
  return (unsigned int)f2h(a) | ((unsigned int)f2h(b) << 16);
}
__device__ __forceinline__ v8f wmma_f16(const Frag& a, const Frag& b, v8f c) {
  return __builtin_amdgcn_wmma_f32_16x16x32_f16(false, a.v, false, b.v,
                                                (short)0, c, false, false);
}

// gfx1250 async global->LDS copy (ASYNCcnt). VDST = raw LDS byte offset
// (low 32 bits of a generic LDS pointer).
// GVS addressing: mem = SADDR64 + VADDR32.
__device__ __forceinline__ void async_ld_b32(unsigned lds_addr,
                                             const void* gbase, unsigned goff) {
  asm volatile("global_load_async_to_lds_b32 %0, %1, %2"
               :: "v"(lds_addr), "v"(goff), "s"(gbase) : "memory");
}
__device__ __forceinline__ void wait_async0() {
  asm volatile("s_wait_asynccnt 0" ::: "memory");
}
__device__ __forceinline__ unsigned lds_off(const void* p) {
  return (unsigned)(size_t)p;
}

// ---------------------------------------------------------------------------
// Generic f16-WMMA GEMM:  Out[M,1280] = rowmap(A)[M,K] * W[K,1280] (+bias+res)
// A rows remapped: src = (r/rpb)*bstride + roff + r%rpb  (identity: rpb huge)
// Double-buffered LDS staging, one barrier per 64-wide K slab.
// A_IS_F16 path stages A via async-to-LDS (no VGPR round trip).
// ---------------------------------------------------------------------------
template <bool A_IS_F16, bool OUT_F16>
__global__ __launch_bounds__(256) void gemm_f16_wmma(
    const void* __restrict__ Av, const float* __restrict__ W,
    const float* __restrict__ bias, const float* __restrict__ resid,
    void* __restrict__ Outv, int M, int K, int lda,
    int rpb, int bstride, int roff)
{
  constexpr int N = kC;
  __shared__ unsigned short As[2][128][66];   // [buf][m][k], stride padded
  __shared__ unsigned short Bs[2][128][66];   // [buf][n][k] (transposed)

  const int tid = threadIdx.x;
  const int lane = tid & 31, wid = tid >> 5;
  const int wm = wid & 3, wn = wid >> 2;      // 4x2 wave grid -> 32x64 / wave
  const int bm = blockIdx.y, bn = blockIdx.x;

  const v8f zv = {0.f,0.f,0.f,0.f,0.f,0.f,0.f,0.f};
  v8f acc[2][4];
  #pragma unroll
  for (int m = 0; m < 2; ++m)
    #pragma unroll
    for (int n = 0; n < 4; ++n) acc[m][n] = zv;

  auto stageA = [&](int k0, int buf) {
    #pragma unroll
    for (int p = 0; p < 8; ++p) {
      int r = p * 16 + (tid >> 4);
      int c = (tid & 15) * 4;
      int grow = bm * 128 + r;
      if (A_IS_F16) {
        if (grow < M) {
          int rb = grow / rpb;
          long src = (long)rb * bstride + roff + (grow - rb * rpb);
          unsigned goff = (unsigned)((src * lda + k0 + c) * 2);
          async_ld_b32(lds_off(&As[buf][r][c]),     Av, goff);
          async_ld_b32(lds_off(&As[buf][r][c + 2]), Av, goff + 4);
        } else {
          *(unsigned int*)&As[buf][r][c]     = 0;
          *(unsigned int*)&As[buf][r][c + 2] = 0;
        }
      } else {
        unsigned int lo = 0, hic = 0;
        if (grow < M) {
          int rb = grow / rpb;
          long src = (long)rb * bstride + roff + (grow - rb * rpb);
          const float* A = (const float*)Av;
          const float4 d = *(const float4*)(A + src * lda + k0 + c);
          lo = pack2(d.x, d.y); hic = pack2(d.z, d.w);
        }
        *(unsigned int*)&As[buf][r][c]     = lo;
        *(unsigned int*)&As[buf][r][c + 2] = hic;
      }
    }
  };
  auto stageB = [&](int k0, int buf) {
    #pragma unroll
    for (int p = 0; p < 4; ++p) {
      int k = p * 16 + (tid >> 5) * 2;      // even k: pack (k, k+1) pairs
      int c = (tid & 31) * 4;
      const float4 d0 = *(const float4*)(W + (long)(k0 + k) * N + bn * 128 + c);
      const float4 d1 = *(const float4*)(W + (long)(k0 + k + 1) * N + bn * 128 + c);
      *(unsigned int*)&Bs[buf][c + 0][k] = pack2(d0.x, d1.x);
      *(unsigned int*)&Bs[buf][c + 1][k] = pack2(d0.y, d1.y);
      *(unsigned int*)&Bs[buf][c + 2][k] = pack2(d0.z, d1.z);
      *(unsigned int*)&Bs[buf][c + 3][k] = pack2(d0.w, d1.w);
    }
  };

  const int nIter = K / 64;
  stageA(0, 0);
  stageB(0, 0);

  for (int i = 0; i < nIter; ++i) {
    if (A_IS_F16) wait_async0();             // async stage of this slab done
    __syncthreads();
    const int buf = i & 1;
    if (i + 1 < nIter) {                     // stage next slab into other buf
      stageA((i + 1) * 64, 1 - buf);
      stageB((i + 1) * 64, 1 - buf);
    }
    if (i + 2 < nIter) {                     // speculative global_prefetch_b8
      int k2 = (i + 2) * 64;
      int r = tid >> 1;
      int grow = bm * 128 + r;
      if (grow < M) {
        int rb = grow / rpb;
        long src = (long)rb * bstride + roff + (grow - rb * rpb);
        if (A_IS_F16)
          __builtin_prefetch((const unsigned short*)Av + src * lda + k2 + (tid & 1) * 32, 0, 1);
        else
          __builtin_prefetch((const float*)Av + src * lda + k2 + (tid & 1) * 32, 0, 1);
      }
      __builtin_prefetch(W + (long)(k2 + (tid >> 2)) * N + bn * 128 + (tid & 3) * 32, 0, 1);
    }
    // ---- two k=32 WMMA steps on current buffer ----
    #pragma unroll
    for (int ks = 0; ks < 2; ++ks) {
      const int kk = ks * 32;
      Frag fa[2], fb[4];
      #pragma unroll
      for (int m = 0; m < 2; ++m) {
        int row = wm * 32 + m * 16 + (lane & 15);
        int kb = kk + ((lane & 16) ? 16 : 0);    // contiguous k = 16h + s (matches B)
        #pragma unroll
        for (int j = 0; j < 4; ++j) {
          fa[m].u[j]     = *(const unsigned int*)&As[buf][row][kb + 2 * j];
          fa[m].u[4 + j] = *(const unsigned int*)&As[buf][row][kb + 8 + 2 * j];
        }
      }
      #pragma unroll
      for (int n = 0; n < 4; ++n) {
        int col = wn * 64 + n * 16 + (lane & 15);
        int kb = kk + ((lane & 16) ? 16 : 0);    // contiguous k = 16h + s (matches A)
        #pragma unroll
        for (int j = 0; j < 8; ++j)
          fb[n].u[j] = *(const unsigned int*)&Bs[buf][col][kb + 2 * j];
      }
      #pragma unroll
      for (int m = 0; m < 2; ++m)
        #pragma unroll
        for (int n = 0; n < 4; ++n)
          acc[m][n] = wmma_f16(fa[m], fb[n], acc[m][n]);
    }
  }
  // ---- epilogue: C/D layout -> VGPR v is row v (lanes 0-15) / v+8 (16-31) ----
  #pragma unroll
  for (int m = 0; m < 2; ++m) {
    int row0 = bm * 128 + wm * 32 + m * 16 + ((lane & 16) ? 8 : 0);
    #pragma unroll
    for (int n = 0; n < 4; ++n) {
      int col = bn * 128 + wn * 64 + n * 16 + (lane & 15);
      #pragma unroll
      for (int v = 0; v < 8; ++v) {
        int grow = row0 + v;
        if (grow < M) {
          float val = acc[m][n][v];
          if (bias)  val += bias[col];
          if (resid) val += resid[(long)grow * N + col];
          if (OUT_F16)
            ((unsigned short*)Outv)[(long)grow * N + col] = f2h(val);
          else
            ((float*)Outv)[(long)grow * N + col] = val;
        }
      }
    }
  }
}

// ---------------------------------------------------------------------------
// Attention: one workgroup per (b,h) x 128 q-rows; 77(+4 ip) keys in LDS.
// K tiles staged via async-to-LDS; V tiles need an element transpose (manual).
// Two independent softmaxes (main + ip) as in the reference, summed into O.
// ---------------------------------------------------------------------------
__global__ __launch_bounds__(256) void attn_f16_wmma(
    const unsigned short* __restrict__ Q,  const unsigned short* __restrict__ Kt,
    const unsigned short* __restrict__ Vt, const unsigned short* __restrict__ Kip,
    const unsigned short* __restrict__ Vip, unsigned short* __restrict__ O)
{
  __shared__ unsigned short Ks [96][66];      // [key][d], keys padded to 96
  __shared__ unsigned short Vs [64][98];      // transposed [d][key]
  __shared__ unsigned short Kis[32][66];      // ip keys padded to 32
  __shared__ unsigned short Vis[64][34];      // transposed [d][ipkey]
  __shared__ unsigned short Ps [8][16][98];   // per-wave probs, K padded 96
  __shared__ unsigned short Pis[8][16][34];   // per-wave ip probs, K padded 32

  const int tid = threadIdx.x, lane = tid & 31, wid = tid >> 5;
  const int bh = blockIdx.y, b = bh / kH, h = bh % kH;
  const int qb = blockIdx.x;
  const int nk = lane & 15;
  const int hl = (lane & 16) ? 1 : 0;

  for (int e = tid; e < kTXT * 32; e += 256) {         // K rows: async copy
    int key = e >> 5, d = (e & 31) * 2;
    unsigned goff = (unsigned)(((b * kTXT + key) * kC + h * kD + d) * 2);
    async_ld_b32(lds_off(&Ks[key][d]), Kt, goff);
  }
  for (int e = tid; e < (96 - kTXT) * 32; e += 256) {  // K pad rows: zero
    int key = kTXT + (e >> 5), d = (e & 31) * 2;
    *(unsigned int*)&Ks[key][d] = 0;
  }
  for (int e = tid; e < kNT * 32; e += 256) {          // Kip rows: async copy
    int key = e >> 5, d = (e & 31) * 2;
    unsigned goff = (unsigned)(((b * kNT + key) * kC + h * kD + d) * 2);
    async_ld_b32(lds_off(&Kis[key][d]), Kip, goff);
  }
  for (int e = tid; e < (32 - kNT) * 32; e += 256) {   // Kip pad rows: zero
    int key = kNT + (e >> 5), d = (e & 31) * 2;
    *(unsigned int*)&Kis[key][d] = 0;
  }
  for (int e = tid; e < 96 * 64; e += 256) {           // stage V (transposed)
    int key = e >> 6, d = e & 63;
    unsigned short v = 0;
    if (key < kTXT) v = Vt[(long)(b * kTXT + key) * kC + h * kD + d];
    Vs[d][key] = v;
  }
  for (int e = tid; e < 32 * 64; e += 256) {           // stage Vip (transposed)
    int key = e >> 6, d = e & 63;
    unsigned short v = 0;
    if (key < kNT) v = Vip[(long)(b * kNT + key) * kC + h * kD + d];
    Vis[d][key] = v;
  }
  for (int e = tid; e < 128; e += 256) {               // zero P pad columns
    int w = e >> 4, r = e & 15;
    #pragma unroll
    for (int c = 80; c < 96; c += 2) *(unsigned int*)&Ps[w][r][c] = 0;
    #pragma unroll
    for (int c = 16; c < 32; c += 2) *(unsigned int*)&Pis[w][r][c] = 0;
  }
  wait_async0();                                       // async K tiles landed
  __syncthreads();

  const int qr = qb * 128 + wid * 16;

  Frag qa[2];                                           // Q fragments (d=64)
  {
    const long qbase = ((long)(b * kSQ) + qr + nk) * kC + h * kD;
    #pragma unroll
    for (int ks = 0; ks < 2; ++ks) {
      int kb = ks * 32 + hl * 16;
      #pragma unroll
      for (int j = 0; j < 4; ++j) {
        qa[ks].u[j]     = *(const unsigned int*)(Q + qbase + kb + 2 * j);
        qa[ks].u[4 + j] = *(const unsigned int*)(Q + qbase + kb + 8 + 2 * j);
      }
    }
  }

  const v8f zv = {0.f,0.f,0.f,0.f,0.f,0.f,0.f,0.f};

  v8f s[5];                                             // S = Q K^T (16x80)
  #pragma unroll
  for (int t = 0; t < 5; ++t) s[t] = zv;
  #pragma unroll
  for (int ks = 0; ks < 2; ++ks) {
    int kb = ks * 32 + hl * 16;
    #pragma unroll
    for (int t = 0; t < 5; ++t) {
      Frag kf;
      int key = t * 16 + nk;
      #pragma unroll
      for (int j = 0; j < 8; ++j)
        kf.u[j] = *(const unsigned int*)&Ks[key][kb + 2 * j];
      s[t] = wmma_f16(qa[ks], kf, s[t]);
    }
  }
  #pragma unroll
  for (int v = 0; v < 8; ++v) {                         // masked softmax (77)
    float e0[5];
    float m = -1e30f;
    #pragma unroll
    for (int t = 0; t < 5; ++t) {
      int key = t * 16 + nk;
      float x = (key < kTXT) ? s[t][v] * kScale : -1e30f;
      e0[t] = x; m = fmaxf(m, x);
    }
    #pragma unroll
    for (int o = 1; o < 16; o <<= 1) m = fmaxf(m, __shfl_xor(m, o, 32));
    float sum = 0.f;
    #pragma unroll
    for (int t = 0; t < 5; ++t) { e0[t] = __expf(e0[t] - m); sum += e0[t]; }
    #pragma unroll
    for (int o = 1; o < 16; o <<= 1) sum += __shfl_xor(sum, o, 32);
    float inv = 1.0f / sum;
    int row = v + hl * 8;
    #pragma unroll
    for (int t = 0; t < 5; ++t)
      Ps[wid][row][t * 16 + nk] = f2h(e0[t] * inv);
  }

  v8f si = zv;                                          // ip branch (4 keys)
  #pragma unroll
  for (int ks = 0; ks < 2; ++ks) {
    int kb = ks * 32 + hl * 16;
    Frag kf;
    #pragma unroll
    for (int j = 0; j < 8; ++j)
      kf.u[j] = *(const unsigned int*)&Kis[nk][kb + 2 * j];
    si = wmma_f16(qa[ks], kf, si);
  }
  #pragma unroll
  for (int v = 0; v < 8; ++v) {
    float x = (nk < kNT) ? si[v] * kScale : -1e30f;
    float m = x;
    #pragma unroll
    for (int o = 1; o < 16; o <<= 1) m = fmaxf(m, __shfl_xor(m, o, 32));
    float e = __expf(x - m);
    float sum = e;
    #pragma unroll
    for (int o = 1; o < 16; o <<= 1) sum += __shfl_xor(sum, o, 32);
    int row = v + hl * 8;
    Pis[wid][row][nk] = f2h(e * (kIpScale / sum));
  }

  v8f o[4];                                             // O = P~V + P~ip Vip
  #pragma unroll
  for (int t = 0; t < 4; ++t) o[t] = zv;
  #pragma unroll
  for (int ks = 0; ks < 3; ++ks) {
    Frag pf;
    int kba = ks * 32 + hl * 16;
    #pragma unroll
    for (int j = 0; j < 4; ++j) {
      pf.u[j]     = *(const unsigned int*)&Ps[wid][nk][kba + 2 * j];
      pf.u[4 + j] = *(const unsigned int*)&Ps[wid][nk][kba + 8 + 2 * j];
    }
    int kbb = ks * 32 + hl * 16;
    #pragma unroll
    for (int t = 0; t < 4; ++t) {
      Frag vf;
      int dcol = t * 16 + nk;
      #pragma unroll
      for (int j = 0; j < 8; ++j)
        vf.u[j] = *(const unsigned int*)&Vs[dcol][kbb + 2 * j];
      o[t] = wmma_f16(pf, vf, o[t]);
    }
  }
  {
    Frag pf;
    int kba = hl * 16;
    #pragma unroll
    for (int j = 0; j < 4; ++j) {
      pf.u[j]     = *(const unsigned int*)&Pis[wid][nk][kba + 2 * j];
      pf.u[4 + j] = *(const unsigned int*)&Pis[wid][nk][kba + 8 + 2 * j];
    }
    int kbb = hl * 16;
    #pragma unroll
    for (int t = 0; t < 4; ++t) {
      Frag vf;
      int dcol = t * 16 + nk;
      #pragma unroll
      for (int j = 0; j < 8; ++j)
        vf.u[j] = *(const unsigned int*)&Vis[dcol][kbb + 2 * j];
      o[t] = wmma_f16(pf, vf, o[t]);
    }
  }
  #pragma unroll
  for (int t = 0; t < 4; ++t) {                         // store O as f16
    #pragma unroll
    for (int v = 0; v < 8; ++v) {
      int row = qr + v + hl * 8;
      long idx = ((long)(b * kSQ) + row) * kC + h * kD + t * 16 + nk;
      O[idx] = f2h(o[t][v]);
    }
  }
}

// ---------------------------------------------------------------------------
extern "C" void kernel_launch(void* const* d_in, const int* in_sizes, int n_in,
                              void* d_out, int out_size, void* d_ws, size_t ws_size,
                              hipStream_t stream) {
  (void)in_sizes; (void)n_in; (void)out_size; (void)ws_size;
  const float* hidden = (const float*)d_in[0];
  const float* enc    = (const float*)d_in[1];
  const float* Wq     = (const float*)d_in[2];
  const float* Wk     = (const float*)d_in[3];
  const float* Wv     = (const float*)d_in[4];
  const float* Wk_ip  = (const float*)d_in[5];
  const float* Wv_ip  = (const float*)d_in[6];
  const float* Wo     = (const float*)d_in[7];
  const float* bo     = (const float*)d_in[8];

  char* ws = (char*)d_ws;
  size_t off = 0;
  auto carve = [&](size_t bytes) {
    void* p = ws + off; off += (bytes + 255) & ~size_t(255); return p;
  };
  unsigned short* Qb   = (unsigned short*)carve((size_t)kB * kSQ  * kC * 2);
  unsigned short* AOb  = (unsigned short*)carve((size_t)kB * kSQ  * kC * 2);
  unsigned short* Kbuf = (unsigned short*)carve((size_t)kB * kTXT * kC * 2);
  unsigned short* Vbuf = (unsigned short*)carve((size_t)kB * kTXT * kC * 2);
  unsigned short* Kib  = (unsigned short*)carve((size_t)kB * kNT  * kC * 2);
  unsigned short* Vib  = (unsigned short*)carve((size_t)kB * kNT  * kC * 2);

  const dim3 blk(256);
  const int IDENT = 1 << 30;
  // Q projection (fp32 in, f16 out)
  gemm_f16_wmma<false, true><<<dim3(10, 256), blk, 0, stream>>>(
      hidden, Wq, nullptr, nullptr, Qb, kB * kSQ, kC, kC, IDENT, 0, 0);
  // K / V projections over txt rows (77 of 81 per batch)
  gemm_f16_wmma<false, true><<<dim3(10, 5), blk, 0, stream>>>(
      enc, Wk, nullptr, nullptr, Kbuf, kB * kTXT, kCAD, kCAD, kTXT, kTXT + kNT, 0);
  gemm_f16_wmma<false, true><<<dim3(10, 5), blk, 0, stream>>>(
      enc, Wv, nullptr, nullptr, Vbuf, kB * kTXT, kCAD, kCAD, kTXT, kTXT + kNT, 0);
  // ip projections over last 4 rows per batch
  gemm_f16_wmma<false, true><<<dim3(10, 1), blk, 0, stream>>>(
      enc, Wk_ip, nullptr, nullptr, Kib, kB * kNT, kCAD, kCAD, kNT, kTXT + kNT, kTXT);
  gemm_f16_wmma<false, true><<<dim3(10, 1), blk, 0, stream>>>(
      enc, Wv_ip, nullptr, nullptr, Vib, kB * kNT, kCAD, kCAD, kNT, kTXT + kNT, kTXT);
  // attention (main + ip), f16 out
  attn_f16_wmma<<<dim3(kSQ / 128, kB * kH), blk, 0, stream>>>(
      Qb, Kbuf, Vbuf, Kib, Vib, AOb);
  // output projection + bias + residual, fp32 out
  gemm_f16_wmma<true, false><<<dim3(10, 256), blk, 0, stream>>>(
      AOb, Wo, bo, hidden, (float*)d_out, kB * kSQ, kC, kC, IDENT, 0, 0);
}
